// DepthWiseSepConv_36945308680627
// MI455X (gfx1250) — matrix operand order repaired and checked by reading, not compiled, not run on hardware
//
#include <hip/hip_runtime.h>

typedef __attribute__((ext_vector_type(16))) __bf16 v16bf;
typedef __attribute__((ext_vector_type(8)))  __bf16 v8bf;
typedef __attribute__((ext_vector_type(8)))  float  v8f;

#define NB     16
#define C_IN   128
#define C_OUT  256
#define HH     112
#define WW     112
#define HWSZ   (HH*WW)          // 12544
#define M_TOTAL (NB*HWSZ)       // 200704
#define BN_EPS 1e-5f

#define MT  128                 // M tile (spatial rows)
#define NT  64                  // N tile (output channels)
#define LDA 136                 // padded K stride (bf16 elems): 272B, 16B aligned

// ---------------------------------------------------------------- utilities
__global__ void zero_kernel(float* p, int n) {
    int i = blockIdx.x * blockDim.x + threadIdx.x;
    if (i < n) p[i] = 0.f;
}

// BN affine params from accumulated sum/sumsq: a = g*rsqrt(var+eps), b = beta - mean*a
__global__ void bnparam_kernel(const float* __restrict__ sum, const float* __restrict__ sq,
                               const float* __restrict__ g, const float* __restrict__ be,
                               float* __restrict__ pa, float* __restrict__ pb,
                               int nch, float inv_cnt) {
    int c = blockIdx.x * blockDim.x + threadIdx.x;
    if (c < nch) {
        float m = sum[c] * inv_cnt;
        float v = sq[c] * inv_cnt - m * m;
        float a = g[c] * __frsqrt_rn(v + BN_EPS);
        pa[c] = a;
        pb[c] = be[c] - m * a;
    }
}

// ------------------------------------------- depthwise 3x3 + bias + BN1 stats
__global__ __launch_bounds__(256)
void dw_kernel(const float* __restrict__ x, const float* __restrict__ w,
               const float* __restrict__ bias, float* __restrict__ y1,
               float* __restrict__ s1sum, float* __restrict__ s1sq) {
    __shared__ float red[4];                       // {sum0, sq0, sum1, sq1}
    const int i0  = blockIdx.x * 256;
    const int i   = i0 + threadIdx.x;
    const int cA  = (i0 / HWSZ) % C_IN;            // a 256-elem block spans <= 2 channels
    const int cB  = ((i0 + 255) / HWSZ) % C_IN;
    if (threadIdx.x < 4) red[threadIdx.x] = 0.f;
    __syncthreads();

    float val = 0.f;
    int   c   = cA;
    if (i < M_TOTAL * C_IN / 1 && i < NB * C_IN * HWSZ) {
        int t  = i;
        int wq = t % WW;  t /= WW;
        int hq = t % HH;  t /= HH;
        c      = t % C_IN; t /= C_IN;
        int b  = t;
        const float* xp = x + (long long)(b * C_IN + c) * HWSZ;
        const float* wp = w + c * 9;
        float acc = bias[c];
#pragma unroll
        for (int dy = -1; dy <= 1; ++dy) {
            int hy = hq + dy;
            if ((unsigned)hy < HH) {
#pragma unroll
                for (int dx = -1; dx <= 1; ++dx) {
                    int wx = wq + dx;
                    if ((unsigned)wx < WW)
                        acc += xp[hy * WW + wx] * wp[(dy + 1) * 3 + (dx + 1)];
                }
            }
        }
        y1[i] = acc;
        val   = acc;
    }

    // wave reduction of (sum, sumsq) split by which of the 2 channels we hit
    int   slot = (c == cA) ? 0 : 1;
    float v0 = slot == 0 ? val : 0.f, q0 = slot == 0 ? val * val : 0.f;
    float v1 = val - v0,              q1 = val * val - q0;
#pragma unroll
    for (int off = 16; off > 0; off >>= 1) {
        v0 += __shfl_down(v0, off); q0 += __shfl_down(q0, off);
        v1 += __shfl_down(v1, off); q1 += __shfl_down(q1, off);
    }
    if ((threadIdx.x & 31) == 0) {
        atomicAdd(&red[0], v0); atomicAdd(&red[1], q0);
        atomicAdd(&red[2], v1); atomicAdd(&red[3], q1);
    }
    __syncthreads();
    if (threadIdx.x == 0) { atomicAdd(&s1sum[cA], red[0]); atomicAdd(&s1sq[cA], red[1]); }
    if (threadIdx.x == 1 && cB != cA) { atomicAdd(&s1sum[cB], red[2]); atomicAdd(&s1sq[cB], red[3]); }
}

// ------------- pointwise 1x1 as WMMA GEMM: BN1+ReLU fused in, bias + BN2 stats out
__global__ __launch_bounds__(256)
void pw_wmma_kernel(const float* __restrict__ y1, const float* __restrict__ pw_w,
                    const float* __restrict__ pw_b, const float* __restrict__ a1,
                    const float* __restrict__ b1, float* __restrict__ out,
                    float* __restrict__ s2sum, float* __restrict__ s2sq) {
    __shared__ __align__(16) char smem[MT * LDA * 2 + NT * LDA * 2]; // A + B (bf16)
    __shared__ float lstat[NT * 2];
    __bf16* la = (__bf16*)smem;                  // [MT][LDA]
    __bf16* lb = (__bf16*)(smem + MT * LDA * 2); // [NT][LDA]
    float*  ld = (float*)smem;                   // D staging [NT][129], aliases A/B

    const int tid = threadIdx.x;
    const int m0  = blockIdx.x * MT;
    const int c0  = blockIdx.y * NT;

    if (tid < NT * 2) lstat[tid] = 0.f;

    // B tile: Bt[n][k] = pw_w[(c0+n)*C_IN + k]  (fully coalesced, fp32 -> bf16)
    for (int idx = tid; idx < NT * C_IN; idx += 256) {
        int n = idx >> 7, k = idx & 127;
        lb[n * LDA + k] = (__bf16)pw_w[(c0 + n) * C_IN + k];
    }
    // A tile with fused BN1+ReLU: a[m][k] = relu(y1[b,k,rem]*a1[k]+b1[k]); coalesced in m
    for (int idx = tid; idx < MT * C_IN; idx += 256) {
        int k = idx >> 7, m = idx & 127;
        int s = m0 + m;
        float v = 0.f;
        if (s < M_TOTAL) {
            int b = s / HWSZ, rem = s - b * HWSZ;
            v = y1[(b * C_IN + k) * HWSZ + rem];
            v = v * a1[k] + b1[k];
            v = v > 0.f ? v : 0.f;
        }
        la[m * LDA + k] = (__bf16)v;
    }
    __syncthreads();

    const int wave  = tid >> 5;
    const int lane  = tid & 31;
    const int lm    = lane & 15;
    const int khalf = lane >> 4;
    const int mrow  = wave * 16 + lm;

    v8f acc[4];
#pragma unroll
    for (int j = 0; j < 4; ++j)
        acc[j] = v8f{0.f, 0.f, 0.f, 0.f, 0.f, 0.f, 0.f, 0.f};

#pragma unroll
    for (int ks = 0; ks < C_IN; ks += 32) {
        // A fragment: 16-bit A layout, K interleaved by lane half (ds_load_b128 x2)
        v16bf af;
        {
            const v8bf alo = *(const v8bf*)&la[mrow * LDA + ks + khalf * 8];
            const v8bf ahi = *(const v8bf*)&la[mrow * LDA + ks + 16 + khalf * 8];
#pragma unroll
            for (int e = 0; e < 8; ++e) { af[e] = alo[e]; af[8 + e] = ahi[e]; }
        }
#pragma unroll
        for (int j = 0; j < 4; ++j) {
            // B fragment: lane = column n, K interleaved by lane half (matches A)
            const int n = j * 16 + lm;
            const v8bf blo = *(const v8bf*)&lb[n * LDA + ks + khalf * 8];
            const v8bf bhi = *(const v8bf*)&lb[n * LDA + ks + 16 + khalf * 8];
            v16bf bfm;
#pragma unroll
            for (int e = 0; e < 8; ++e) { bfm[e] = blo[e]; bfm[8 + e] = bhi[e]; }
            acc[j] = __builtin_amdgcn_wmma_f32_16x16x32_bf16(
                false, af, false, bfm, (short)0, acc[j], false, false);
        }
    }
    __syncthreads();   // done reading la/lb; reuse region for D staging

    // epilogue: bias add, BN2 stats, stage D tile (transposed) in LDS
#pragma unroll
    for (int j = 0; j < 4; ++j) {
        const int nl   = j * 16 + lm;
        const float bb = pw_b[c0 + nl];
        float lsum = 0.f, lsq = 0.f;
#pragma unroll
        for (int r = 0; r < 8; ++r) {
            const int ml = wave * 16 + khalf * 8 + r;   // C/D layout: lanes 16-31 -> M+8
            float v = acc[j][r] + bb;
            ld[nl * 129 + ml] = v;
            if (m0 + ml < M_TOTAL) { lsum += v; lsq += v * v; }
        }
        lsum += __shfl_xor(lsum, 16);                   // fold the two lane-halves (same n)
        lsq  += __shfl_xor(lsq, 16);
        if (khalf == 0) {
            atomicAdd(&lstat[nl * 2 + 0], lsum);
            atomicAdd(&lstat[nl * 2 + 1], lsq);
        }
    }
    __syncthreads();

    // coalesced global store of pre-BN2 activations
    for (int idx = tid; idx < NT * MT; idx += 256) {
        int n = idx >> 7, m = idx & 127;
        int s = m0 + m;
        if (s < M_TOTAL) {
            int b = s / HWSZ, rem = s - b * HWSZ;
            out[(b * C_OUT + c0 + n) * HWSZ + rem] = ld[n * 129 + m];
        }
    }
    if (tid < NT) {
        atomicAdd(&s2sum[c0 + tid], lstat[tid * 2 + 0]);
        atomicAdd(&s2sq[c0 + tid],  lstat[tid * 2 + 1]);
    }
}

// ----------------------------------------------- BN2 + ReLU in-place (float4)
__global__ __launch_bounds__(256)
void bn2_apply_kernel(float4* __restrict__ out, const float* __restrict__ pa,
                      const float* __restrict__ pb, int n4) {
    int i = blockIdx.x * 256 + threadIdx.x;
    if (i < n4) {
        int e = i * 4;
        int c = (e / HWSZ) % C_OUT;                     // HWSZ % 4 == 0 -> uniform per float4
        float a = pa[c], b = pb[c];
        float4 v = out[i];
        v.x = fmaxf(v.x * a + b, 0.f);
        v.y = fmaxf(v.y * a + b, 0.f);
        v.z = fmaxf(v.z * a + b, 0.f);
        v.w = fmaxf(v.w * a + b, 0.f);
        out[i] = v;
    }
}

// --------------------------------------------------------------------- launch
extern "C" void kernel_launch(void* const* d_in, const int* in_sizes, int n_in,
                              void* d_out, int out_size, void* d_ws, size_t ws_size,
                              hipStream_t stream) {
    const float* x    = (const float*)d_in[0];
    const float* dw_w = (const float*)d_in[1];
    const float* dw_b = (const float*)d_in[2];
    const float* pw_w = (const float*)d_in[3];
    const float* pw_b = (const float*)d_in[4];
    const float* bn1g = (const float*)d_in[5];
    const float* bn1b = (const float*)d_in[6];
    const float* bn2g = (const float*)d_in[7];
    const float* bn2b = (const float*)d_in[8];
    (void)in_sizes; (void)n_in; (void)ws_size;

    const long long NSf = (long long)NB * C_IN * HWSZ;   // 25,690,112 floats
    float* ws    = (float*)d_ws;
    float* y1    = ws;
    float* s1sum = ws + NSf;
    float* s1sq  = s1sum + C_IN;
    float* p1a   = s1sq + C_IN;
    float* p1b   = p1a + C_IN;
    float* s2sum = p1b + C_IN;
    float* s2sq  = s2sum + C_OUT;
    float* p2a   = s2sq + C_OUT;
    float* p2b   = p2a + C_OUT;

    const int statN = 4 * C_IN + 4 * C_OUT;              // everything after y1
    zero_kernel<<<(statN + 255) / 256, 256, 0, stream>>>(s1sum, statN);

    const int dwBlocks = (int)((NSf + 255) / 256);       // 100,352
    dw_kernel<<<dwBlocks, 256, 0, stream>>>(x, dw_w, dw_b, y1, s1sum, s1sq);

    const float inv_cnt = 1.0f / (float)M_TOTAL;
    bnparam_kernel<<<1, 128, 0, stream>>>(s1sum, s1sq, bn1g, bn1b, p1a, p1b, C_IN, inv_cnt);

    dim3 g(M_TOTAL / MT, C_OUT / NT);                    // (1568, 4)
    pw_wmma_kernel<<<g, 256, 0, stream>>>(y1, pw_w, pw_b, p1a, p1b,
                                          (float*)d_out, s2sum, s2sq);

    bnparam_kernel<<<1, 256, 0, stream>>>(s2sum, s2sq, bn2g, bn2b, p2a, p2b, C_OUT, inv_cnt);

    const int n4 = (NB * C_OUT * HWSZ) / 4;              // 12,845,056
    bn2_apply_kernel<<<(n4 + 255) / 256, 256, 0, stream>>>((float4*)d_out, p2a, p2b, n4);
}
